// GraphConsistencyModel_90589450207804
// MI455X (gfx1250) — matrix operand order repaired and checked by reading, not compiled, not run on hardware
//
#include <hip/hip_runtime.h>
#include <hip/hip_bf16.h>
#include <math.h>

// ---------------------------------------------------------------------------
// GraphConsistencyModel on gfx1250 (MI455X): bf16 WMMA GEMM pipeline.
// D = A x B + C via v_wmma_f32_16x16x32_bf16, wave32.
// Wave-level 4x4 register blocking: each wave owns a 64x64 output tile
// (16 v8f accumulators); per K-step: 4 A-frags + 4 B-frags -> 16 WMMAs.
// Block = 8 waves = 128 rows x 256 cols.
// ---------------------------------------------------------------------------

typedef __bf16 bf16_t;
typedef __attribute__((ext_vector_type(16))) __bf16 v16bf;
typedef __attribute__((ext_vector_type(8)))  __bf16 v8bf;
typedef __attribute__((ext_vector_type(8)))  float  v8f;

#define NN    50000
#define EE    800000
#define GG    10000
#define LMM   768
#define HH    256
#define EDIMM 64

// -------------------- WMMA fragment helpers --------------------------------
// A-frag (16x32 bf16, ISA 7.12.2): lane L<16 holds row L, K = {0..7,16..23};
// lane L+16 holds row L, K = {8..15,24..31}.
__device__ __forceinline__ v16bf cat8(v8bf lo, v8bf hi) {
  return __builtin_shufflevector(lo, hi, 0,1,2,3,4,5,6,7,8,9,10,11,12,13,14,15);
}
__device__ __forceinline__ v16bf load_a_bf(const bf16_t* __restrict__ row, int kb) {
  v8bf lo = *(const v8bf*)(row + kb);        // K = kb .. kb+7
  v8bf hi = *(const v8bf*)(row + kb + 16);   // K = kb+16 .. kb+23
  return cat8(lo, hi);
}
__device__ __forceinline__ v16bf load_a_f32(const float* __restrict__ row, int kb) {
  v8f flo = *(const v8f*)(row + kb);
  v8f fhi = *(const v8f*)(row + kb + 16);
  return cat8(__builtin_convertvector(flo, v8bf), __builtin_convertvector(fhi, v8bf));
}
#define WMMA_BF16(A, B, C) \
  __builtin_amdgcn_wmma_f32_16x16x32_bf16(false, (A), false, (B), (short)0, (C), false, false)

#define ACC_INIT(acc)                                                   \
  _Pragma("unroll") for (int _n = 0; _n < 16; ++_n)                     \
    acc[_n] = v8f{0.f,0.f,0.f,0.f,0.f,0.f,0.f,0.f};

// Load 4 B-frags for this kt / wave-column-group, then 16 WMMAs.
#define WMMA_4X4_STEP(a, Bbase, kt, ct, acc)                            \
  {                                                                     \
    v16bf _b[4];                                                        \
    _Pragma("unroll") for (int _j = 0; _j < 4; ++_j)                    \
      _b[_j] = (Bbase)[(size_t)((kt) * 16 + (ct) * 4 + _j) * 32];       \
    _Pragma("unroll") for (int _rt = 0; _rt < 4; ++_rt)                 \
      _Pragma("unroll") for (int _j = 0; _j < 4; ++_j)                  \
        acc[_rt * 4 + _j] = WMMA_BF16(a[_rt], _b[_j], acc[_rt * 4 + _j]); \
  }

// -------------------- weight repack: f32 row-major -> bf16 B-fragments -----
// B-frag (32x16 bf16): lane l<16 -> col l, K = {0..7,16..23}; lane l>=16 -> col l-16,
// K = {8..15,24..31}  (v16bf element i packs K pairs low/high per VGPR).
__global__ __launch_bounds__(256) void repack_kernel(
    const float* __restrict__ W, bf16_t* __restrict__ out, int K, int N) {
  int idx = blockIdx.x * 256 + threadIdx.x;
  if (idx >= K * N) return;
  int i    = idx & 15;
  int lane = (idx >> 4) & 31;
  int frag = idx >> 9;
  int NT   = N >> 4;
  int kt = frag / NT, nt = frag % NT;
  int k = kt * 32 + ((lane >> 4) << 3) + (i & 7) + ((i >> 3) << 4);
  int n = nt * 16 + (lane & 15);
  out[idx] = (bf16_t)W[(size_t)k * N + n];
}

// -------------------- req-dependent bias vectors ---------------------------
__global__ __launch_bounds__(256) void rbias_kernel(
    const float* __restrict__ req, const float* __restrict__ W_req,
    const float* __restrict__ node_W1, const float* __restrict__ node_b1,
    const float* __restrict__ gap_W1,  const float* __restrict__ gap_b1,
    float* __restrict__ req_proj, float* __restrict__ node_rbias,
    float* __restrict__ gap_rbias) {
  int j = threadIdx.x;
  if (blockIdx.x == 0) {
    float s = 0.f;
    for (int k = 0; k < LMM; ++k) s += req[k] * W_req[(size_t)k * HH + j];
    req_proj[j] = fmaxf(s, 0.f);
  } else if (blockIdx.x == 1) {
    float s = node_b1[j];
    for (int k = 0; k < LMM; ++k) s += req[k] * node_W1[(size_t)(HH + k) * HH + j];
    node_rbias[j] = s;
  } else {
    float s = gap_b1[j];
    for (int k = 0; k < LMM; ++k) s += req[k] * gap_W1[(size_t)(576 + k) * HH + j];
    gap_rbias[j] = s;
  }
}

// -------------------- h0 = node_feats + req_proj (store bf16) --------------
__global__ __launch_bounds__(256) void h0_kernel(
    const float* __restrict__ node_feats, const float* __restrict__ req_proj,
    bf16_t* __restrict__ h0, int total) {
  int i = blockIdx.x * 256 + threadIdx.x;
  if (i < total) h0[i] = (bf16_t)(node_feats[i] + req_proj[i & (HH - 1)]);
}

// -------------------- message GEMM + relu + atomic scatter -----------------
// (E,320)x(320,256): A row e = [h[src[e]] (bf16), edge_feats[e] (f32->bf16)]
__global__ __launch_bounds__(256) void msg_gemm_scatter(
    const bf16_t* __restrict__ hbf, const int* __restrict__ src_idx,
    const int* __restrict__ dst_idx, const float* __restrict__ edge_feats,
    const bf16_t* __restrict__ Wfrag, const float* __restrict__ bias,
    float* __restrict__ agg, int nEdges) {
  const int wave = threadIdx.x >> 5;
  const int lane = threadIdx.x & 31;
  const int ct = wave >> 1;                         // column group (64 cols)
  const int rowBase = blockIdx.x * 128 + (wave & 1) * 64;  // 64 rows / wave
  const int r = lane & 15;
  const int kHalf = (lane >> 4) * 8;

  const bf16_t* aH[4];
  const float*  aE[4];
#pragma unroll
  for (int rt = 0; rt < 4; ++rt) {
    int e = rowBase + rt * 16 + r;
    if (e >= nEdges) e = nEdges - 1;
    aH[rt] = hbf + (size_t)src_idx[e] * HH;
    aE[rt] = edge_feats + (size_t)e * EDIMM;
  }
  const v16bf* Bbase = (const v16bf*)Wfrag + lane;

  v8f acc[16]; ACC_INIT(acc);
#pragma unroll
  for (int kt = 0; kt < 10; ++kt) {
    v16bf a[4];
#pragma unroll
    for (int rt = 0; rt < 4; ++rt)
      a[rt] = (kt < 8) ? load_a_bf (aH[rt], kt * 32 + kHalf)
                       : load_a_f32(aE[rt], (kt - 8) * 32 + kHalf);
    WMMA_4X4_STEP(a, Bbase, kt, ct, acc);
  }

  // D layout: acc[rt*4+j][v] = row (rt*16 + rsel + v), col ct*64 + j*16 + dcol
  const int dcol = lane & 15;
  const int rsel = (lane < 16) ? 0 : 8;
#pragma unroll
  for (int rt = 0; rt < 4; ++rt) {
    int dsts[8]; bool ok[8];
#pragma unroll
    for (int v = 0; v < 8; ++v) {
      int grow = rowBase + rt * 16 + rsel + v;
      ok[v]   = grow < nEdges;
      dsts[v] = dst_idx[ok[v] ? grow : 0];
    }
#pragma unroll
    for (int j = 0; j < 4; ++j) {
      int col = ct * 64 + j * 16 + dcol;
      float bb = bias[col];
#pragma unroll
      for (int v = 0; v < 8; ++v) {
        if (ok[v]) {
          float m = acc[rt * 4 + j][v] + bb;
          m = m > 0.f ? m : 0.f;
          atomicAdd(agg + (size_t)dsts[v] * HH + col, m);
        }
      }
    }
  }
}

// -------------------- update GEMM: relu([h,agg] @ Wu + bu) -> bf16 ---------
__global__ __launch_bounds__(256) void update_gemm(
    const bf16_t* __restrict__ hin, const float* __restrict__ agg,
    const bf16_t* __restrict__ Wfrag, const float* __restrict__ bias,
    bf16_t* __restrict__ hout, int nRows) {
  const int wave = threadIdx.x >> 5;
  const int lane = threadIdx.x & 31;
  const int ct = wave >> 1;
  const int rowBase = blockIdx.x * 128 + (wave & 1) * 64;
  const int r = lane & 15;
  const int kHalf = (lane >> 4) * 8;

  const bf16_t* aH[4];
  const float*  aA[4];
#pragma unroll
  for (int rt = 0; rt < 4; ++rt) {
    int m = rowBase + rt * 16 + r;
    if (m >= nRows) m = nRows - 1;
    aH[rt] = hin + (size_t)m * HH;
    aA[rt] = agg + (size_t)m * HH;
  }
  const v16bf* Bbase = (const v16bf*)Wfrag + lane;

  v8f acc[16]; ACC_INIT(acc);
#pragma unroll
  for (int kt = 0; kt < 16; ++kt) {
    v16bf a[4];
#pragma unroll
    for (int rt = 0; rt < 4; ++rt)
      a[rt] = (kt < 8) ? load_a_bf (aH[rt], kt * 32 + kHalf)
                       : load_a_f32(aA[rt], (kt - 8) * 32 + kHalf);
    WMMA_4X4_STEP(a, Bbase, kt, ct, acc);
  }

  const int dcol = lane & 15;
  const int rsel = (lane < 16) ? 0 : 8;
#pragma unroll
  for (int rt = 0; rt < 4; ++rt)
#pragma unroll
    for (int j = 0; j < 4; ++j) {
      int col = ct * 64 + j * 16 + dcol;
      float bb = bias[col];
#pragma unroll
      for (int v = 0; v < 8; ++v) {
        int grow = rowBase + rt * 16 + rsel + v;
        if (grow < nRows) {
          float x = acc[rt * 4 + j][v] + bb;
          x = x > 0.f ? x : 0.f;
          hout[(size_t)grow * HH + col] = (bf16_t)x;
        }
      }
    }
}

// -------------------- node_h = h @ W_out + b_out (f32 + bf16 copy) ---------
__global__ __launch_bounds__(256) void final_proj(
    const bf16_t* __restrict__ hin, const bf16_t* __restrict__ Wfrag,
    const float* __restrict__ bias, float* __restrict__ nodeh_f32,
    bf16_t* __restrict__ nodeh_bf, int nRows) {
  const int wave = threadIdx.x >> 5;
  const int lane = threadIdx.x & 31;
  const int ct = wave >> 1;
  const int rowBase = blockIdx.x * 128 + (wave & 1) * 64;
  const int r = lane & 15;
  const int kHalf = (lane >> 4) * 8;

  const bf16_t* aH[4];
#pragma unroll
  for (int rt = 0; rt < 4; ++rt) {
    int m = rowBase + rt * 16 + r;
    if (m >= nRows) m = nRows - 1;
    aH[rt] = hin + (size_t)m * HH;
  }
  const v16bf* Bbase = (const v16bf*)Wfrag + lane;

  v8f acc[16]; ACC_INIT(acc);
#pragma unroll
  for (int kt = 0; kt < 8; ++kt) {
    v16bf a[4];
#pragma unroll
    for (int rt = 0; rt < 4; ++rt) a[rt] = load_a_bf(aH[rt], kt * 32 + kHalf);
    WMMA_4X4_STEP(a, Bbase, kt, ct, acc);
  }

  const int dcol = lane & 15;
  const int rsel = (lane < 16) ? 0 : 8;
#pragma unroll
  for (int rt = 0; rt < 4; ++rt)
#pragma unroll
    for (int j = 0; j < 4; ++j) {
      int col = ct * 64 + j * 16 + dcol;
      float bb = bias[col];
#pragma unroll
      for (int v = 0; v < 8; ++v) {
        int grow = rowBase + rt * 16 + rsel + v;
        if (grow < nRows) {
          float x = acc[rt * 4 + j][v] + bb;
          nodeh_f32[(size_t)grow * HH + col] = x;
          nodeh_bf [(size_t)grow * HH + col] = (bf16_t)x;
        }
      }
    }
}

// -------------------- node head hidden: relu(tool_h @ W1a + rbias) ---------
__global__ __launch_bounds__(256) void node_hidden_gemm(
    const bf16_t* __restrict__ abase /* = nodeh_bf + HH (skip node 0) */,
    const bf16_t* __restrict__ Wfrag, const float* __restrict__ rbias,
    float* __restrict__ hidden, int nRows) {
  const int wave = threadIdx.x >> 5;
  const int lane = threadIdx.x & 31;
  const int ct = wave >> 1;
  const int rowBase = blockIdx.x * 128 + (wave & 1) * 64;
  const int r = lane & 15;
  const int kHalf = (lane >> 4) * 8;

  const bf16_t* aH[4];
#pragma unroll
  for (int rt = 0; rt < 4; ++rt) {
    int m = rowBase + rt * 16 + r;
    if (m >= nRows) m = nRows - 1;
    aH[rt] = abase + (size_t)m * HH;
  }
  const v16bf* Bbase = (const v16bf*)Wfrag + lane;

  v8f acc[16]; ACC_INIT(acc);
#pragma unroll
  for (int kt = 0; kt < 8; ++kt) {
    v16bf a[4];
#pragma unroll
    for (int rt = 0; rt < 4; ++rt) a[rt] = load_a_bf(aH[rt], kt * 32 + kHalf);
    WMMA_4X4_STEP(a, Bbase, kt, ct, acc);
  }

  const int dcol = lane & 15;
  const int rsel = (lane < 16) ? 0 : 8;
#pragma unroll
  for (int rt = 0; rt < 4; ++rt)
#pragma unroll
    for (int j = 0; j < 4; ++j) {
      int col = ct * 64 + j * 16 + dcol;
      float bb = rbias[col];
#pragma unroll
      for (int v = 0; v < 8; ++v) {
        int grow = rowBase + rt * 16 + rsel + v;
        if (grow < nRows) {
          float x = acc[rt * 4 + j][v] + bb;
          hidden[(size_t)grow * HH + col] = x > 0.f ? x : 0.f;
        }
      }
    }
}

// -------------------- gap head hidden: relu([hu,hv,ef] @ W1a + rbias) ------
__global__ __launch_bounds__(256) void gap_hidden_gemm(
    const bf16_t* __restrict__ nodeh_bf, const int* __restrict__ gap_u,
    const int* __restrict__ gap_v, const float* __restrict__ gap_ef,
    const bf16_t* __restrict__ Wfrag, const float* __restrict__ rbias,
    float* __restrict__ hidden, int nRows) {
  const int wave = threadIdx.x >> 5;
  const int lane = threadIdx.x & 31;
  const int ct = wave >> 1;
  const int rowBase = blockIdx.x * 128 + (wave & 1) * 64;
  const int r = lane & 15;
  const int kHalf = (lane >> 4) * 8;

  const bf16_t* aU[4];
  const bf16_t* aV[4];
  const float*  aE[4];
#pragma unroll
  for (int rt = 0; rt < 4; ++rt) {
    int g = rowBase + rt * 16 + r;
    if (g >= nRows) g = nRows - 1;
    aU[rt] = nodeh_bf + (size_t)gap_u[g] * HH;
    aV[rt] = nodeh_bf + (size_t)gap_v[g] * HH;
    aE[rt] = gap_ef + (size_t)g * EDIMM;
  }
  const v16bf* Bbase = (const v16bf*)Wfrag + lane;

  v8f acc[16]; ACC_INIT(acc);
#pragma unroll
  for (int kt = 0; kt < 18; ++kt) {
    v16bf a[4];
#pragma unroll
    for (int rt = 0; rt < 4; ++rt) {
      if (kt < 8)       a[rt] = load_a_bf (aU[rt], kt * 32 + kHalf);
      else if (kt < 16) a[rt] = load_a_bf (aV[rt], (kt - 8)  * 32 + kHalf);
      else              a[rt] = load_a_f32(aE[rt], (kt - 16) * 32 + kHalf);
    }
    WMMA_4X4_STEP(a, Bbase, kt, ct, acc);
  }

  const int dcol = lane & 15;
  const int rsel = (lane < 16) ? 0 : 8;
#pragma unroll
  for (int rt = 0; rt < 4; ++rt)
#pragma unroll
    for (int j = 0; j < 4; ++j) {
      int col = ct * 64 + j * 16 + dcol;
      float bb = rbias[col];
#pragma unroll
      for (int v = 0; v < 8; ++v) {
        int grow = rowBase + rt * 16 + rsel + v;
        if (grow < nRows) {
          float x = acc[rt * 4 + j][v] + bb;
          hidden[(size_t)grow * HH + col] = x > 0.f ? x : 0.f;
        }
      }
    }
}

// -------------------- row dot W2 + b2 -> logit, sigmoid -> risk ------------
__global__ __launch_bounds__(256) void head_reduce(
    const float* __restrict__ hidden, const float* __restrict__ W2,
    const float* __restrict__ b2, float* __restrict__ logits,
    float* __restrict__ risks, int nRows) {
  int wave = threadIdx.x >> 5, lane = threadIdx.x & 31;
  int row = blockIdx.x * 8 + wave;
  if (row >= nRows) return;
  const float* hr = hidden + (size_t)row * HH;
  float s = 0.f;
#pragma unroll
  for (int i = 0; i < 8; ++i) { int c = lane * 8 + i; s += hr[c] * W2[c]; }
  for (int off = 16; off; off >>= 1) s += __shfl_xor(s, off, 32);
  if (lane == 0) {
    float lg = s + b2[0];
    logits[row] = lg;
    risks[row]  = 1.f / (1.f + expf(-lg));
  }
}

// -------------------- tool_h mean/max pooling ------------------------------
__global__ __launch_bounds__(256) void meanmax_part(
    const float* __restrict__ node_h, float* __restrict__ psum,
    float* __restrict__ pmax) {
  int t = threadIdx.x, b = blockIdx.x;
  float s = 0.f, m = -INFINITY;
  for (int row = 1 + b; row < NN; row += 256) {
    float v = node_h[(size_t)row * HH + t];
    s += v; m = fmaxf(m, v);
  }
  psum[b * HH + t] = s;
  pmax[b * HH + t] = m;
}
__global__ __launch_bounds__(256) void meanmax_final(
    const float* __restrict__ psum, const float* __restrict__ pmax,
    float* __restrict__ hG) {
  int t = threadIdx.x;
  float s = 0.f, m = -INFINITY;
  for (int b = 0; b < 256; ++b) { s += psum[b * HH + t]; m = fmaxf(m, pmax[b * HH + t]); }
  hG[t]      = s / (float)(NN - 1);
  hG[HH + t] = m;
}

// -------------------- consistency head (single row) ------------------------
__global__ __launch_bounds__(256) void cons_kernel(
    const float* __restrict__ hG, const float* __restrict__ req,
    const float* __restrict__ W1, const float* __restrict__ b1,
    const float* __restrict__ W2, const float* __restrict__ b2,
    float* __restrict__ out01) {
  __shared__ float red[256];
  int j = threadIdx.x;
  float s = b1[j];
  for (int k = 0; k < 2 * HH; ++k) s += hG[k] * W1[(size_t)k * HH + j];
  for (int k = 0; k < LMM; ++k)    s += req[k] * W1[(size_t)(2 * HH + k) * HH + j];
  red[j] = fmaxf(s, 0.f) * W2[j];
  __syncthreads();
  for (int st = 128; st; st >>= 1) { if (j < st) red[j] += red[j + st]; __syncthreads(); }
  if (j == 0) {
    float lg = red[0] + b2[0];
    out01[1] = lg;
    out01[0] = 1.f / (1.f + expf(-lg));
  }
}

// ---------------------------------------------------------------------------
extern "C" void kernel_launch(void* const* d_in, const int* in_sizes, int n_in,
                              void* d_out, int out_size, void* d_ws, size_t ws_size,
                              hipStream_t stream) {
  const float* node_feats = (const float*)d_in[0];
  const int*   edge_index = (const int*)d_in[1];
  const float* req_emb    = (const float*)d_in[2];
  const float* edge_feats = (const float*)d_in[3];
  const int*   gap_u      = (const int*)d_in[4];
  const int*   gap_v      = (const int*)d_in[5];
  const float* gap_ef     = (const float*)d_in[6];
  const float* W_req      = (const float*)d_in[7];
  const float* Wm         = (const float*)d_in[8];
  const float* bm         = (const float*)d_in[9];
  const float* Wu         = (const float*)d_in[10];
  const float* bu         = (const float*)d_in[11];
  const float* W_out      = (const float*)d_in[12];
  const float* b_out      = (const float*)d_in[13];
  const float* cons_W1    = (const float*)d_in[14];
  const float* cons_b1    = (const float*)d_in[15];
  const float* cons_W2    = (const float*)d_in[16];
  const float* cons_b2    = (const float*)d_in[17];
  const float* node_W1    = (const float*)d_in[18];
  const float* node_b1    = (const float*)d_in[19];
  const float* node_W2    = (const float*)d_in[20];
  const float* node_b2    = (const float*)d_in[21];
  const float* gap_W1     = (const float*)d_in[22];
  const float* gap_b1     = (const float*)d_in[23];
  const float* gap_W2     = (const float*)d_in[24];
  const float* gap_b2     = (const float*)d_in[25];

  const int* src_idx = edge_index;
  const int* dst_idx = edge_index + EE;

  // ----- output layout (flat, reference return order) -----
  float* out            = (float*)d_out;
  float* out_S          = out;                     // S(1), S_logit(1)
  float* out_node_risks = out + 2;                 // 49999
  float* out_node_logit = out + 50001;             // 49999
  float* out_gap_risks  = out + 100000;            // 10000
  float* out_gap_logit  = out + 110000;            // 10000
  float* out_node_h     = out + 120000;            // 50000*256
  float* out_hG         = out + 12920000;          // 512
  (void)in_sizes; (void)n_in; (void)out_size;

  // ----- workspace carving -----
  char* p = (char*)d_ws;
  size_t off = 0;
  auto carve = [&](size_t bytes) -> char* {
    char* r = p + off;
    off = (off + bytes + 255) & ~(size_t)255;
    return r;
  };
  float*  req_proj   = (float*)carve(HH * 4);
  float*  node_rbias = (float*)carve(HH * 4);
  float*  gap_rbias  = (float*)carve(HH * 4);
  float*  psum       = (float*)carve(256 * HH * 4);
  float*  pmax       = (float*)carve(256 * HH * 4);
  bf16_t* hA         = (bf16_t*)carve((size_t)NN * HH * 2);
  bf16_t* hB         = (bf16_t*)carve((size_t)NN * HH * 2);
  bf16_t* nodeh_bf   = (bf16_t*)carve((size_t)NN * HH * 2);
  float*  agg        = (float*)carve((size_t)NN * HH * 4);  // reused as head hidden
  bf16_t* WmF        = (bf16_t*)carve((size_t)3 * 320 * HH * 2);
  bf16_t* WuF        = (bf16_t*)carve((size_t)3 * 512 * HH * 2);
  bf16_t* WoutF      = (bf16_t*)carve((size_t)HH * HH * 2);
  bf16_t* nW1F       = (bf16_t*)carve((size_t)HH * HH * 2);
  bf16_t* gW1F       = (bf16_t*)carve((size_t)576 * HH * 2);
  (void)ws_size;

  // ----- 1. repack all weights into bf16 WMMA B-fragment layout -----
  for (int l = 0; l < 3; ++l) {
    repack_kernel<<<(320 * HH + 255) / 256, 256, 0, stream>>>(
        Wm + (size_t)l * 320 * HH, WmF + (size_t)l * 320 * HH, 320, HH);
    repack_kernel<<<(512 * HH + 255) / 256, 256, 0, stream>>>(
        Wu + (size_t)l * 512 * HH, WuF + (size_t)l * 512 * HH, 512, HH);
  }
  repack_kernel<<<(HH * HH + 255) / 256, 256, 0, stream>>>(W_out, WoutF, HH, HH);
  repack_kernel<<<(HH * HH + 255) / 256, 256, 0, stream>>>(node_W1, nW1F, HH, HH);
  repack_kernel<<<(576 * HH + 255) / 256, 256, 0, stream>>>(gap_W1, gW1F, 576, HH);

  // ----- 2. req-dependent bias vectors (folds concat(x, req) @ W1) -----
  rbias_kernel<<<3, 256, 0, stream>>>(req_emb, W_req, node_W1, node_b1,
                                      gap_W1, gap_b1, req_proj, node_rbias, gap_rbias);

  // ----- 3. h0 = node_feats + relu(req @ W_req), stored bf16 -----
  h0_kernel<<<(NN * HH + 255) / 256, 256, 0, stream>>>(node_feats, req_proj, hA, NN * HH);

  // ----- 4. three GNN layers: msg GEMM + atomic scatter, update GEMM -----
  bf16_t* hcur = hA;
  bf16_t* hnxt = hB;
  for (int l = 0; l < 3; ++l) {
    hipMemsetAsync(agg, 0, (size_t)NN * HH * sizeof(float), stream);
    msg_gemm_scatter<<<EE / 128, 256, 0, stream>>>(
        hcur, src_idx, dst_idx, edge_feats,
        WmF + (size_t)l * 320 * HH, bm + (size_t)l * HH, agg, EE);
    update_gemm<<<(NN + 127) / 128, 256, 0, stream>>>(
        hcur, agg, WuF + (size_t)l * 512 * HH, bu + (size_t)l * HH, hnxt, NN);
    bf16_t* t = hcur; hcur = hnxt; hnxt = t;
  }

  // ----- 5. node_h = h @ W_out + b_out (output #7) -----
  final_proj<<<(NN + 127) / 128, 256, 0, stream>>>(
      hcur, WoutF, b_out, out_node_h, nodeh_bf, NN);

  // ----- 6. h_G = [mean, max] over tool_h (output #8) -----
  meanmax_part<<<256, 256, 0, stream>>>(out_node_h, psum, pmax);
  meanmax_final<<<1, 256, 0, stream>>>(psum, pmax, out_hG);

  // ----- 7. consistency head (outputs #1, #2) -----
  cons_kernel<<<1, 256, 0, stream>>>(out_hG, req_emb, cons_W1, cons_b1,
                                     cons_W2, cons_b2, out_S);

  // ----- 8. node head (outputs #3, #4); hidden reuses agg buffer -----
  node_hidden_gemm<<<(NN - 1 + 127) / 128, 256, 0, stream>>>(
      nodeh_bf + HH, nW1F, node_rbias, agg, NN - 1);
  head_reduce<<<(NN - 1 + 7) / 8, 256, 0, stream>>>(
      agg, node_W2, node_b2, out_node_logit, out_node_risks, NN - 1);

  // ----- 9. gap head (outputs #5, #6) -----
  gap_hidden_gemm<<<(GG + 127) / 128, 256, 0, stream>>>(
      nodeh_bf, gap_u, gap_v, gap_ef, gW1F, gap_rbias, agg, GG);
  head_reduce<<<(GG + 7) / 8, 256, 0, stream>>>(
      agg, gap_W2, gap_b2, out_gap_logit, out_gap_risks, GG);
}
